// QuantizedLinear_27977416966777
// MI455X (gfx1250) — matrix operand order repaired and checked by reading, not compiled, not run on hardware
//
#include <hip/hip_runtime.h>

typedef _Float16 v16h __attribute__((ext_vector_type(16)));
typedef _Float16 v8h  __attribute__((ext_vector_type(8)));
typedef _Float16 v4h  __attribute__((ext_vector_type(4)));
typedef _Float16 v2h  __attribute__((ext_vector_type(2)));
typedef float    v8f  __attribute__((ext_vector_type(8)));
typedef unsigned int u32x4 __attribute__((ext_vector_type(4)));
typedef int i32x8 __attribute__((ext_vector_type(8)));
typedef int i32x4 __attribute__((ext_vector_type(4)));

#define TOKENS 2048
#define KDIM   4096
#define NDIM   11008
#define BM     128
#define BN     256
#define BK     32
#define LDA    40                 // A row stride in f16 (80B), bank-friendly
#define BROW   48                 // B raw row stride in bytes (32 data + 16 pad)
#define NK     (KDIM / BK)        // 128 K-steps

#define A_BUF_BYTES  (BM * LDA * 2)        // 10240
#define A_BYTES      (2 * A_BUF_BYTES)     // 20480
#define B_BUF_BYTES  (BN * BROW)           // 12288
#define SMEM_BYTES   (A_BYTES + 2 * B_BUF_BYTES)  // 45056

#if __has_builtin(__builtin_amdgcn_tensor_load_to_lds) && __has_builtin(__builtin_amdgcn_s_wait_tensorcnt)
#define HAVE_TDM 1
#else
#define HAVE_TDM 0
#endif

__device__ __forceinline__ v16h cat16(v8h a, v8h b) {
    return __builtin_shufflevector(a, b, 0,1,2,3,4,5,6,7,8,9,10,11,12,13,14,15);
}

// 16 raw u8 weights (one b128 LDS read) -> 16 dequantized f16.
// f16 trick: as_f16(0x6400 | q) == 1024 + q exactly; w = (1024+q)*s + (-(1024+zp)*s)
__device__ __forceinline__ v16h dq_frag(const unsigned char* p, v2h s2, v2h m2) {
    uint4 r = *(const uint4*)p;
    unsigned int u[4] = { r.x, r.y, r.z, r.w };
    v16h out;
#pragma unroll
    for (int i = 0; i < 4; ++i) {
        unsigned lo = __builtin_amdgcn_perm(0x64646464u, u[i], 0x04010400u); // {64,b1,64,b0}
        unsigned hi = __builtin_amdgcn_perm(0x64646464u, u[i], 0x04030402u); // {64,b3,64,b2}
        v2h wlo = __builtin_bit_cast(v2h, lo) * s2 + m2;   // v_pk_fma_f16
        v2h whi = __builtin_bit_cast(v2h, hi) * s2 + m2;
        out[i*4+0] = wlo[0]; out[i*4+1] = wlo[1];
        out[i*4+2] = whi[0]; out[i*4+3] = whi[1];
    }
    return out;
}

__global__ void __launch_bounds__(256)
qlinear_wmma_kernel(const float* __restrict__ X,
                    const unsigned char* __restrict__ Q,
                    const float* __restrict__ S,
                    const float* __restrict__ Z,
                    const float* __restrict__ Bias,
                    float* __restrict__ Out)
{
    __shared__ __attribute__((aligned(16))) unsigned char smem[SMEM_BYTES];
    _Float16* Ah = (_Float16*)smem;                  // 2 x [BM][LDA] f16

    const int tid  = threadIdx.x;
    const int lane = tid & 31;
    const int wave = tid >> 5;
    const int wy   = wave >> 2;     // 0..1 : M group (64 rows)
    const int wx   = wave & 3;      // 0..3 : N group (64 cols)
    const int half = lane >> 4;     // 0/1
    const int l16  = lane & 15;

    const int m0 = blockIdx.y * BM;
    const int n0 = blockIdx.x * BN;

    // ---- X staging: 128 rows x 32 f32 = 4 float4 / thread ------------------
    const int xr = tid >> 3;
    const int xc = (tid & 7) * 4;
    const float* xbase = X + (long)m0 * KDIM;
    float4 xst[4];

    // ---- per-lane dequant constants (lane owns 4 output channels) ----------
    v2h s2[4], m2[4];
#pragma unroll
    for (int nt = 0; nt < 4; ++nt) {
        const int ch = n0 + wx * 64 + nt * 16 + l16;
        const float s = S[ch];
        const float m = -(1024.0f + Z[ch]) * s;
        s2[nt][0] = (_Float16)s;  s2[nt][1] = (_Float16)s;
        m2[nt][0] = (_Float16)m;  m2[nt][1] = (_Float16)m;
    }

    const unsigned char* qtile = Q + (long)n0 * KDIM;

#if HAVE_TDM
    // One TDM descriptor load per K-step: 256 rows x 32B tile, tensor stride 4096,
    // LDS padding 4 dwords per 8 dwords -> 48B row stride.
    auto tdm_issue = [&](int buf, int kk) {
        const unsigned long ga = (unsigned long)(qtile + kk);
        u32x4 g0;
        g0[0] = 1u;                                        // count=1
        g0[1] = (unsigned)(A_BYTES + buf * B_BUF_BYTES);   // lds_addr
        g0[2] = (unsigned)ga;                              // global_addr[31:0]
        g0[3] = ((unsigned)(ga >> 32) & 0x01FFFFFFu) | (2u << 30); // addr[56:32] | type=2
        i32x8 g1;
        g1[0] = (1 << 20) | (2 << 22) | (3 << 25);  // pad_enable, interval=8dw, amount=4dw
        g1[1] = (int)((unsigned)KDIM << 16);        // tensor_dim0 = 4096 (low16 in [31:16])
        g1[2] = (int)((unsigned)NDIM << 16);        // tensor_dim0 hi=0 | tensor_dim1 low16
        g1[3] = (int)(32u << 16);                   // tensor_dim1 hi=0 | tile_dim0=32
        g1[4] = 256;                                // tile_dim1=256, tile_dim2=0
        g1[5] = KDIM;                               // tensor_dim0_stride = 4096
        g1[6] = 0;
        g1[7] = 0;
        i32x4 z4 = (i32x4){0, 0, 0, 0};
        i32x8 z8 = (i32x8){0, 0, 0, 0, 0, 0, 0, 0};
        __builtin_amdgcn_tensor_load_to_lds(g0, g1, z4, z4, z8, 0);
    };
#else
    // Fallback: staged per-thread raw u8 copy (2 x 16B chunks)
    uint4 bst0, bst1;
    const int br = tid >> 1;             // rows tid>>1 and tid>>1 + 128
    const int bsg = (tid & 1) * 16;
    const unsigned char* qb0 = qtile + (long)br * KDIM + bsg;
    const unsigned char* qb1 = qtile + (long)(br + 128) * KDIM + bsg;
#endif

    auto load_stage = [&](int kk) {
#pragma unroll
        for (int p = 0; p < 4; ++p)
            xst[p] = *(const float4*)(xbase + (long)(p * 32 + xr) * KDIM + kk + xc);
#if !HAVE_TDM
        bst0 = *(const uint4*)(qb0 + kk);
        bst1 = *(const uint4*)(qb1 + kk);
#endif
    };

    auto store_stage = [&](int buf) {
#pragma unroll
        for (int p = 0; p < 4; ++p) {
            v4h h;
            h[0] = (_Float16)xst[p].x;  h[1] = (_Float16)xst[p].y;
            h[2] = (_Float16)xst[p].z;  h[3] = (_Float16)xst[p].w;
            *(v4h*)&Ah[buf * (BM * LDA) + (p * 32 + xr) * LDA + xc] = h;
        }
#if !HAVE_TDM
        unsigned char* bb = smem + A_BYTES + buf * B_BUF_BYTES;
        *(uint4*)(bb + br * BROW + bsg)         = bst0;
        *(uint4*)(bb + (br + 128) * BROW + bsg) = bst1;
#endif
    };

    v8f acc[4][4];
#pragma unroll
    for (int mt = 0; mt < 4; ++mt)
#pragma unroll
        for (int nt = 0; nt < 4; ++nt)
            acc[mt][nt] = (v8f){0.f,0.f,0.f,0.f,0.f,0.f,0.f,0.f};

    // ---- pipeline prologue -------------------------------------------------
#if HAVE_TDM
    if (wave == 0) tdm_issue(0, 0);
#endif
    load_stage(0);
    store_stage(0);
#if HAVE_TDM
    if (wave == 0) __builtin_amdgcn_s_wait_tensorcnt(0);
#endif
    __syncthreads();

    // ---- main K loop -------------------------------------------------------
    for (int kt = 0; kt < NK; ++kt) {
        const int cur = kt & 1;
        if (kt + 1 < NK) {
#if HAVE_TDM
            if (wave == 0) tdm_issue(cur ^ 1, (kt + 1) * BK);
#endif
            load_stage((kt + 1) * BK);
        }

        // B fragments: raw u8 from LDS + perm/pk_fma dequant (shared across mt)
        const unsigned char* bbase = smem + A_BYTES + cur * B_BUF_BYTES + half * 16;
        v16h bfrag[4];
#pragma unroll
        for (int nt = 0; nt < 4; ++nt) {
            const int col = wx * 64 + nt * 16 + l16;
            bfrag[nt] = dq_frag(bbase + col * BROW, s2[nt], m2[nt]);
        }

        // Stream A fragments (contiguous k = 16*half + s, matching B)
        const _Float16* abase = Ah + cur * (BM * LDA);
#pragma unroll
        for (int mt = 0; mt < 4; ++mt) {
            const int row = wy * 64 + mt * 16 + l16;
            const _Float16* pa = abase + row * LDA + half * 16;
            const v16h af = cat16(*(const v8h*)pa, *(const v8h*)(pa + 8));
#pragma unroll
            for (int nt = 0; nt < 4; ++nt)
                acc[mt][nt] = __builtin_amdgcn_wmma_f32_16x16x32_f16(
                    false, af, false, bfrag[nt],
                    (short)0, acc[mt][nt], false, false);
        }

        if (kt + 1 < NK) store_stage(cur ^ 1);
#if HAVE_TDM
        if (wave == 0) __builtin_amdgcn_s_wait_tensorcnt(0);
#endif
        __syncthreads();
    }

    // ---- epilogue: acc + bias ---------------------------------------------
#pragma unroll
    for (int nt = 0; nt < 4; ++nt) {
        const int o  = n0 + wx * 64 + nt * 16 + l16;
        const float bv = Bias[o];
#pragma unroll
        for (int mt = 0; mt < 4; ++mt) {
            const int tbase = m0 + wy * 64 + mt * 16 + half * 8;
#pragma unroll
            for (int j = 0; j < 8; ++j)
                Out[(long)(tbase + j) * NDIM + o] = acc[mt][nt][j] + bv;
        }
    }
}

extern "C" void kernel_launch(void* const* d_in, const int* in_sizes, int n_in,
                              void* d_out, int out_size, void* d_ws, size_t ws_size,
                              hipStream_t stream) {
    const float*         x    = (const float*)d_in[0];
    const unsigned char* qw   = (const unsigned char*)d_in[1];
    const float*         sc   = (const float*)d_in[2];
    const float*         zp   = (const float*)d_in[3];
    const float*         bias = (const float*)d_in[4];
    float*               out  = (float*)d_out;

    dim3 grid(NDIM / BN, TOKENS / BM);   // (43, 16)
    dim3 block(256);
    qlinear_wmma_kernel<<<grid, block, 0, stream>>>(x, qw, sc, zp, bias, out);
}
